// Attention_4252017623134
// MI455X (gfx1250) — compile-verified, matrix operand order checked by reading, not run on hardware
//
#include <hip/hip_runtime.h>
#include <cstdint>
#include <cstddef>

typedef __attribute__((ext_vector_type(16))) _Float16 v16h;
typedef __attribute__((ext_vector_type(8)))  float    v8f;
typedef _Float16 half_t;

#define DIMSZ   1024
#define NSEQ    1024
#define BATCH   8
#define NHEADS  16
#define DHEAD   64
#define INNERD  1024
#define NKPAD   1056    /* 1 null + 1024 keys + 31 pad rows */
#define NKREAL  1025
#define ALPHA_F 128.0f
#define SCALE_F (0.125f / 128.0f)   /* dhead^-0.5 / alpha */
#define EPS_F   1e-5f

__device__ __forceinline__ v8f wmma_f16(v16h a, v16h b, v8f c) {
  // D = A(16x32 f16) * B(32x16 f16) + C(16x16 f32)
  return __builtin_amdgcn_wmma_f32_16x16x32_f16(false, a, false, b, (short)0, c, false, false);
}

__device__ __forceinline__ v16h ld16h(const half_t* p) { return *(const v16h*)p; }

// ---------------------------------------------------------------- fp32 -> fp16
__global__ void cvt_f2h(const float* __restrict__ s, half_t* __restrict__ d, int n) {
  int i = blockIdx.x * blockDim.x + threadIdx.x;
  int stride = gridDim.x * blockDim.x;
  for (; i < n; i += stride) d[i] = (half_t)s[i];
}

// ------------------------------------------------- input layernorm + fp16 stage
// one block per row of 1024; writes raw x (context) and normalized xn, both fp16
__global__ __launch_bounds__(256) void ln_in(const float* __restrict__ x,
                                             const float* __restrict__ gin,
                                             half_t* __restrict__ xh,
                                             half_t* __restrict__ xnh) {
  __shared__ float s_red[16];
  __shared__ float s_stat[2];
  const int tid = threadIdx.x, w = tid >> 5, lane = tid & 31;
  const float* row = x + (size_t)blockIdx.x * DIMSZ;
  float sum = 0.f, ss = 0.f;
  for (int c = tid; c < DIMSZ; c += 256) { float v = row[c]; sum += v; ss += v * v; }
  for (int o = 16; o >= 1; o >>= 1) {
    sum += __shfl_xor(sum, o, 32);
    ss  += __shfl_xor(ss, o, 32);
  }
  if (lane == 0) { s_red[w] = sum; s_red[8 + w] = ss; }
  __syncthreads();
  if (tid == 0) {
    float S = 0.f, Q = 0.f;
    for (int i = 0; i < 8; ++i) { S += s_red[i]; Q += s_red[8 + i]; }
    float mean = S * (1.0f / DIMSZ);
    float var  = Q * (1.0f / DIMSZ) - mean * mean;
    s_stat[0] = mean;
    s_stat[1] = rsqrtf(var + EPS_F);
  }
  __syncthreads();
  const float mean = s_stat[0], inv = s_stat[1];
  half_t* xr  = xh  + (size_t)blockIdx.x * DIMSZ;
  half_t* xnr = xnh + (size_t)blockIdx.x * DIMSZ;
  for (int c = tid; c < DIMSZ; c += 256) {
    float v = row[c];
    xr[c]  = (half_t)v;
    xnr[c] = (half_t)((v - mean) * inv * gin[c]);
  }
}

// ---------------------- null-token rows/cols + zero padding (k row-major, v^T)
__global__ void init_kv(const float* __restrict__ nullkv,
                        half_t* __restrict__ kb, half_t* __restrict__ vt) {
  const int bb = blockIdx.x;
  const int d = threadIdx.x;           // 64 threads
  const size_t kbase = (size_t)bb * NKPAD * DHEAD;
  const size_t vbase = (size_t)bb * DHEAD * NKPAD;
  kb[kbase + d] = (half_t)nullkv[d];                 // k row 0 = null_k
  vt[vbase + (size_t)d * NKPAD + 0] = (half_t)nullkv[DHEAD + d];  // vT col 0 = null_v
  for (int r = NKREAL; r < NKPAD; ++r) {
    kb[kbase + (size_t)r * DHEAD + d] = (half_t)0.f; // pad key rows
    vt[vbase + (size_t)d * NKPAD + r] = (half_t)0.f; // pad vT cols
  }
}

// --------------------------------------------- q = (xn @ Wq^T) * SCALE  (fp16)
// 64 rows per block (4 M-tiles per wave) to amortize weight reads 4x.
// output layout: q[((b*H + h)*N + i)*64 + d]
__global__ __launch_bounds__(256) void gemm_q(const half_t* __restrict__ xn,
                                              const half_t* __restrict__ Wq,
                                              half_t* __restrict__ q) {
  const int tid = threadIdx.x, w = tid >> 5, lane = tid & 31;
  const int lh = lane >> 4, ln = lane & 15;
  const int gi0 = blockIdx.x * 64;
  const int bb = gi0 / NSEQ, ii0 = gi0 % NSEQ;
  const half_t* arow[4];
#pragma unroll
  for (int mt = 0; mt < 4; ++mt)
    arow[mt] = xn + (size_t)(gi0 + mt * 16 + ln) * DIMSZ;
  for (int t = 0; t < 8; ++t) {
    const int nt = w + t * 8;                       // 64 N-tiles over 8 waves
    const half_t* brow = Wq + (size_t)(nt * 16 + ln) * DIMSZ;
    v8f acc[4] = {{}, {}, {}, {}};
    for (int ks = 0; ks < 32; ++ks) {
      const int ko = ks * 32 + 16 * lh;
      const v16h b = ld16h(brow + ko);
#pragma unroll
      for (int mt = 0; mt < 4; ++mt)
        acc[mt] = wmma_f16(ld16h(arow[mt] + ko), b, acc[mt]);
    }
    const int j = nt * 16 + ln;
    const int hh = j >> 6, dd = j & 63;
    const size_t base = ((size_t)bb * NHEADS + hh) * NSEQ;
#pragma unroll
    for (int mt = 0; mt < 4; ++mt)
      for (int r = 0; r < 8; ++r) {
        const int m = mt * 16 + lh * 8 + r;
        q[(base + ii0 + m) * DHEAD + dd] = (half_t)(acc[mt][r] * SCALE_F);
      }
  }
}

// ----------------------------------------------------- kv = x @ Wkv^T  (fp16)
// k rows land at kb[b][1+i][d]; v is stored TRANSPOSED: vt[b][d][1+i]
__global__ __launch_bounds__(256) void gemm_kv(const half_t* __restrict__ xr,
                                               const half_t* __restrict__ Wkv,
                                               half_t* __restrict__ kb,
                                               half_t* __restrict__ vt) {
  const int tid = threadIdx.x, w = tid >> 5, lane = tid & 31;
  const int lh = lane >> 4, ln = lane & 15;
  const int gi0 = blockIdx.x * 16;
  const int bb = gi0 / NSEQ, ii0 = gi0 % NSEQ;
  const half_t* arow = xr + (size_t)(gi0 + ln) * DIMSZ;
  const int nt = w;                                  // 8 N-tiles, 1 per wave
  const half_t* brow = Wkv + (size_t)(nt * 16 + ln) * DIMSZ;
  v8f c = {};
  for (int ks = 0; ks < 32; ++ks) {
    const int ko = ks * 32 + 16 * lh;
    c = wmma_f16(ld16h(arow + ko), ld16h(brow + ko), c);
  }
  const int j = nt * 16 + ln;                        // 0..127
  for (int r = 0; r < 8; ++r) {
    const int m = lh * 8 + r;
    const int row = ii0 + m + 1;                     // +1: null token at 0
    if (j < DHEAD)
      kb[((size_t)bb * NKPAD + row) * DHEAD + j] = (half_t)c[r];
    else
      vt[((size_t)bb * DHEAD + (j - DHEAD)) * NKPAD + row] = (half_t)c[r];
  }
}

// ------------------------------------------------------------------- attention
// grid (N/16, HEADS, B); 128 threads = 4 waves per workgroup.
// phase1: logits = ALPHA * (q.k) with masks -> fp32 LDS strip 16x1056
//         (tiles fully above the causal diagonal skip the WMMAs)
// phase2: row softmax (fp32, single exp pass) -> fp16 probability strip
// phase3: out = P @ V^T via WMMA over the causally-live K range
__global__ __launch_bounds__(128) void attn_kernel(const half_t* __restrict__ q,
                                                   const half_t* __restrict__ kb,
                                                   const half_t* __restrict__ vt,
                                                   const unsigned char* __restrict__ mask,
                                                   half_t* __restrict__ ao) {
  __shared__ __align__(32) float  s_sim[16 * NKPAD];  // 67.5 KB
  __shared__ __align__(32) half_t s_p[16 * NKPAD];    // 33.8 KB
  const int tid = threadIdx.x, w = tid >> 5, lane = tid & 31;
  const int lh = lane >> 4, ln = lane & 15;
  const int bb = blockIdx.z, hh = blockIdx.y, i0 = blockIdx.x * 16;
  const int jmax = i0 + 16;                 // largest visible key col (i0+15+1)

  const half_t* qbase = q + (((size_t)bb * NHEADS + hh) * NSEQ + i0) * DHEAD;
  const half_t* kbase = kb + (size_t)bb * NKPAD * DHEAD;
  const unsigned char* mb = mask + (size_t)bb * NSEQ;

  // q rows for this tile are loop-invariant
  const half_t* arow = qbase + (size_t)ln * DHEAD;
  const v16h a0 = ld16h(arow + 16 * lh);
  const v16h a1 = ld16h(arow + 32 + 16 * lh);

  // ---- phase 1: QK^T + masks
  for (int jt = w; jt < NKPAD / 16; jt += 4) {
    const int jbase = jt * 16;
    const int jcol = jbase + ln;
    if (jbase > jmax) {
      // entire tile above causal diagonal: all -inf, no math
      for (int r = 0; r < 8; ++r)
        s_sim[(lh * 8 + r) * NKPAD + jcol] = -__builtin_huge_valf();
      continue;
    }
    const half_t* brow = kbase + (size_t)(jbase + ln) * DHEAD;
    v8f c = {};
    c = wmma_f16(a0, ld16h(brow + 16 * lh), c);
    c = wmma_f16(a1, ld16h(brow + 32 + 16 * lh), c);
    const bool pm = (jcol == 0) ? true
                  : (jcol < NKREAL ? (mb[jcol - 1] != 0) : false);
    for (int r = 0; r < 8; ++r) {
      const int m = lh * 8 + r;
      const int irow = i0 + m;                     // local query index
      const bool vis = pm && (jcol <= irow + 1);   // causal: mask j >= i+2
      s_sim[m * NKPAD + jcol] = vis ? c[r] * ALPHA_F : -__builtin_huge_valf();
    }
  }
  __syncthreads();

  // ---- phase 2: softmax, 8 lanes per row, 132 cols per lane, one exp pass
  {
    const int row = w * 4 + (lane >> 3);
    const int chunk = lane & 7;
    const int CW = NKPAD / 8;  // 132
    float* sr = s_sim + row * NKPAD + chunk * CW;
    float mx = -__builtin_huge_valf();
    for (int c2 = 0; c2 < CW; ++c2) mx = fmaxf(mx, sr[c2]);
    for (int o = 4; o >= 1; o >>= 1) mx = fmaxf(mx, __shfl_xor(mx, o, 8));
    float sum = 0.f;
    for (int c2 = 0; c2 < CW; ++c2) {
      const float e = __expf(sr[c2] - mx);
      sr[c2] = e;                        // store exp in place (lane-exclusive)
      sum += e;
    }
    for (int o = 4; o >= 1; o >>= 1) sum += __shfl_xor(sum, o, 8);
    const float inv = 1.0f / sum;
    half_t* pr = s_p + row * NKPAD + chunk * CW;
    for (int c2 = 0; c2 < CW; ++c2) pr[c2] = (half_t)(sr[c2] * inv);
  }
  __syncthreads();

  // ---- phase 3: P @ V^T (one 16-wide d-tile per wave, causally-live K only)
  {
    const int dbase = w * 16;
    const half_t* vrow = vt + ((size_t)bb * DHEAD + dbase + ln) * NKPAD;
    const int nks = jmax / 32 + 1;                  // K-steps with nonzero P
    v8f c = {};
    for (int ks = 0; ks < nks; ++ks) {
      const int ko = ks * 32 + 16 * lh;
      const v16h a = *(const v16h*)(s_p + ln * NKPAD + ko);
      const v16h b = ld16h(vrow + ko);              // contiguous: V stored ^T
      c = wmma_f16(a, b, c);
    }
    const int dd = dbase + ln;
    for (int r = 0; r < 8; ++r) {
      const int m = lh * 8 + r;
      ao[((size_t)bb * NSEQ + i0 + m) * INNERD + hh * DHEAD + dd] = (half_t)c[r];
    }
  }
}

// -------------------------------- o = ao @ Wo^T, fused final layernorm -> fp32
// 64 rows per block (4 M-tiles per wave); 128 KB fp16 staging in LDS.
__global__ __launch_bounds__(256) void gemm_o_ln(const half_t* __restrict__ ao,
                                                 const half_t* __restrict__ Wo,
                                                 const float* __restrict__ gout,
                                                 float* __restrict__ out) {
  __shared__ __align__(32) half_t s_o[64 * INNERD];  // 128 KB
  const int tid = threadIdx.x, w = tid >> 5, lane = tid & 31;
  const int lh = lane >> 4, ln = lane & 15;
  const int gi0 = blockIdx.x * 64;
  const half_t* arow[4];
#pragma unroll
  for (int mt = 0; mt < 4; ++mt)
    arow[mt] = ao + (size_t)(gi0 + mt * 16 + ln) * INNERD;
  for (int t = 0; t < 8; ++t) {
    const int nt = w + t * 8;
    const half_t* brow = Wo + (size_t)(nt * 16 + ln) * INNERD;
    v8f acc[4] = {{}, {}, {}, {}};
    for (int ks = 0; ks < 32; ++ks) {
      const int ko = ks * 32 + 16 * lh;
      const v16h b = ld16h(brow + ko);
#pragma unroll
      for (int mt = 0; mt < 4; ++mt)
        acc[mt] = wmma_f16(ld16h(arow[mt] + ko), b, acc[mt]);
    }
#pragma unroll
    for (int mt = 0; mt < 4; ++mt)
      for (int r = 0; r < 8; ++r) {
        const int m = mt * 16 + lh * 8 + r;
        s_o[m * INNERD + nt * 16 + ln] = (half_t)acc[mt][r];
      }
  }
  __syncthreads();
  // layernorm: wave w handles rows 8w..8w+7; 4 lanes per row, 256 cols per lane
  const int row = w * 8 + (lane >> 2);
  const int q4 = lane & 3;
  const half_t* sr = s_o + row * INNERD;
  float sum = 0.f, ss = 0.f;
  for (int c2 = q4 * 256; c2 < q4 * 256 + 256; ++c2) {
    const float v = (float)sr[c2];
    sum += v; ss += v * v;
  }
  for (int o = 2; o >= 1; o >>= 1) {
    sum += __shfl_xor(sum, o, 4);
    ss  += __shfl_xor(ss, o, 4);
  }
  const float mean = sum * (1.0f / INNERD);
  const float var  = ss * (1.0f / INNERD) - mean * mean;
  const float inv  = rsqrtf(var + EPS_F);
  float* orow = out + (size_t)(gi0 + row) * DIMSZ;
  for (int c2 = q4 * 256; c2 < q4 * 256 + 256; ++c2)
    orow[c2] = ((float)sr[c2] - mean) * inv * gout[c2];
}

// ------------------------------------------------------------------- launcher
extern "C" void kernel_launch(void* const* d_in, const int* in_sizes, int n_in,
                              void* d_out, int out_size, void* d_ws, size_t ws_size,
                              hipStream_t stream) {
  (void)in_sizes; (void)n_in; (void)out_size; (void)ws_size;
  const float*         x      = (const float*)d_in[0];
  const unsigned char* mask   = (const unsigned char*)d_in[1];
  const float*         g_in   = (const float*)d_in[2];
  const float*         Wq     = (const float*)d_in[3];
  const float*         Wkv    = (const float*)d_in[4];
  const float*         nullkv = (const float*)d_in[5];
  const float*         Wo     = (const float*)d_in[6];
  const float*         g_out  = (const float*)d_in[7];
  float*               out    = (float*)d_out;

  char* ws = (char*)d_ws;
  size_t off = 0;
  auto take = [&](size_t nhalfs) -> half_t* {
    half_t* p = (half_t*)(ws + off);
    off += ((nhalfs * sizeof(half_t) + 255) & ~(size_t)255);
    return p;
  };
  half_t* Wq_h  = take((size_t)INNERD * DIMSZ);
  half_t* Wo_h  = take((size_t)DIMSZ * INNERD);
  half_t* Wkv_h = take((size_t)2 * DHEAD * DIMSZ);
  half_t* x_h   = take((size_t)BATCH * NSEQ * DIMSZ);
  half_t* xn_h  = take((size_t)BATCH * NSEQ * DIMSZ);
  half_t* q_h   = take((size_t)BATCH * NHEADS * NSEQ * DHEAD);
  half_t* k_h   = take((size_t)BATCH * NKPAD * DHEAD);
  half_t* vt_h  = take((size_t)BATCH * DHEAD * NKPAD);
  half_t* ao_h  = xn_h;  // xn dead after gemm_q; reuse for attention output

  cvt_f2h<<<1024, 256, 0, stream>>>(Wq,  Wq_h,  INNERD * DIMSZ);
  cvt_f2h<<<1024, 256, 0, stream>>>(Wo,  Wo_h,  DIMSZ * INNERD);
  cvt_f2h<<<512,  256, 0, stream>>>(Wkv, Wkv_h, 2 * DHEAD * DIMSZ);
  ln_in<<<BATCH * NSEQ, 256, 0, stream>>>(x, g_in, x_h, xn_h);
  init_kv<<<BATCH, DHEAD, 0, stream>>>(nullkv, k_h, vt_h);
  gemm_q<<<BATCH * NSEQ / 64, 256, 0, stream>>>(xn_h, Wq_h, q_h);
  gemm_kv<<<BATCH * NSEQ / 16, 256, 0, stream>>>(x_h, Wkv_h, k_h, vt_h);
  attn_kernel<<<dim3(NSEQ / 16, NHEADS, BATCH), 128, 0, stream>>>(q_h, k_h, vt_h, mask, ao_h);
  gemm_o_ln<<<BATCH * NSEQ / 64, 256, 0, stream>>>(ao_h, Wo_h, g_out, out);
}
